// RJ_63024350102313
// MI455X (gfx1250) — matrix operand order repaired and checked by reading, not compiled, not run on hardware
//
#include <hip/hip_runtime.h>

typedef __bf16 bf16;
typedef __attribute__((ext_vector_type(16))) __bf16 v16bf;
typedef __attribute__((ext_vector_type(8)))  __bf16 v8bf;
typedef __attribute__((ext_vector_type(8)))  float  v8f;

#define D_DIM 512
#define L_SEQ 2048
#define NROWS 16384   // B*L

static __device__ __forceinline__ bf16 f2bf(float f) {
    union { float f; unsigned u; } v; v.f = f;
    unsigned r = (v.u + 0x7FFFu + ((v.u >> 16) & 1u)) >> 16;
    union { unsigned short s; bf16 b; } o; o.s = (unsigned short)r;
    return o.b;
}

// A-fragment (16x32 bf16): lane m = lane&15; K elems {koff..koff+7, koff+16..koff+23}, koff = (lane>=16)*8
static __device__ __forceinline__ v16bf mk_afrag(const bf16* base, int koff) {
    v8bf lo = *(const v8bf*)(base + koff);
    v8bf hi = *(const v8bf*)(base + koff + 16);
    return __builtin_shufflevector(lo, hi, 0,1,2,3,4,5,6,7,8,9,10,11,12,13,14,15);
}
// B-fragment (32x16 bf16): lane n = lane&15; K elems {khalf/2..khalf/2+7, khalf/2+16..khalf/2+23}, khalf = (lane>=16)*16
static __device__ __forceinline__ v16bf mk_bfrag(const bf16* base, int khalf) {
    v8bf lo = *(const v8bf*)(base + khalf / 2);
    v8bf hi = *(const v8bf*)(base + khalf / 2 + 16);
    return __builtin_shufflevector(lo, hi, 0,1,2,3,4,5,6,7,8,9,10,11,12,13,14,15);
}

static __device__ __forceinline__ v8f wmma_bf16(v16bf a, v16bf b, v8f c) {
    return __builtin_amdgcn_wmma_f32_16x16x32_bf16(false, a, false, b, (short)0, c, false, false);
}

// ---------------- Kernel 1: Q/K/V projection GEMM (x @ W), bf16 out ----------------
// MODE 0/1: row-major store (Q/K). MODE 2: transposed store (V -> Vt[d][row]).
template <int MODE>
__global__ __launch_bounds__(256) void qkv_gemm_t(
    const float* __restrict__ x, const float* __restrict__ W, bf16* __restrict__ dst)
{
    __shared__ bf16 As[128 * 32];   // [row][k]
    __shared__ bf16 Bs[128 * 32];   // [n][k] (transposed W tile)
    const int rBase = blockIdx.x * 128;
    const int nBase = blockIdx.y * 128;
    const int tid   = threadIdx.x;
    const int lane  = tid & 31;
    const int lane15 = lane & 15;
    const int wave  = tid >> 5;
    const int waveM = wave >> 1, waveN = wave & 1;
    const int koffA  = (lane >= 16) ? 8 : 0;
    const int khalfB = (lane >= 16) ? 16 : 0;

    v8f zero = {};
    v8f acc[2][4];
    #pragma unroll
    for (int i = 0; i < 2; ++i)
        #pragma unroll
        for (int j = 0; j < 4; ++j) acc[i][j] = zero;

    for (int dk = 0; dk < D_DIM; dk += 32) {
        __syncthreads();
        { // stage x tile [128][32] -> bf16, packed ds_store_b128
            int row = tid >> 1, half = tid & 1;
            const float* src = x + (size_t)(rBase + row) * D_DIM + dk + half * 16;
            v8bf t0, t1;
            #pragma unroll
            for (int j = 0; j < 8; ++j) t0[j] = f2bf(src[j]);
            #pragma unroll
            for (int j = 0; j < 8; ++j) t1[j] = f2bf(src[j + 8]);
            *(v8bf*)(As + row * 32 + half * 16)     = t0;
            *(v8bf*)(As + row * 32 + half * 16 + 8) = t1;
        }
        { // stage W tile [32][128] transposed -> Bs[n][k]
            int k = tid >> 3, seg = tid & 7;
            const float* src = W + (size_t)(dk + k) * D_DIM + nBase + seg * 16;
            #pragma unroll
            for (int j = 0; j < 16; ++j) Bs[(seg * 16 + j) * 32 + k] = f2bf(src[j]);
        }
        __syncthreads();

        v16bf a[2], bb[4];
        #pragma unroll
        for (int i = 0; i < 2; ++i)
            a[i] = mk_afrag(As + (waveM * 32 + i * 16 + lane15) * 32, koffA);
        #pragma unroll
        for (int j = 0; j < 4; ++j)
            bb[j] = mk_bfrag(Bs + (waveN * 64 + j * 16 + lane15) * 32, khalfB);
        #pragma unroll
        for (int i = 0; i < 2; ++i)
            #pragma unroll
            for (int j = 0; j < 4; ++j)
                acc[i][j] = wmma_bf16(a[i], bb[j], acc[i][j]);
    }

    const int hi8 = (lane >= 16) ? 8 : 0;
    #pragma unroll
    for (int i = 0; i < 2; ++i)
        #pragma unroll
        for (int j = 0; j < 4; ++j) {
            int col = nBase + waveN * 64 + j * 16 + lane15;
            #pragma unroll
            for (int r = 0; r < 8; ++r) {
                int row = rBase + waveM * 32 + i * 16 + r + hi8;
                bf16 v = f2bf(acc[i][j][r]);
                if (MODE < 2) dst[(size_t)row * D_DIM + col] = v;     // Q / K
                else          dst[(size_t)col * NROWS + row] = v;     // V transposed
            }
        }
}

// ---------------- Kernel 2: masked flash attention ----------------
// 64 Q rows per block, 4 waves x 16 exclusive rows; O accumulator in LDS (f32).
__global__ __launch_bounds__(128) void attn_kernel(
    const bf16* __restrict__ Qb, const bf16* __restrict__ Kb,
    const bf16* __restrict__ Vt, const int* __restrict__ adj,
    bf16* __restrict__ Ob)
{
    extern __shared__ char smem_raw[];
    float* O     = (float*)smem_raw;                 // 64*512 f32 = 128KB
    bf16*  Pst   = (bf16*)(O + 64 * D_DIM);          // 4 waves * 16*32 bf16
    float* corrS = (float*)(Pst + 4 * 16 * 32);      // 64
    float* invlS = corrS + 64;                       // 64

    const int tid   = threadIdx.x;
    const int lane  = tid & 31;
    const int wave  = tid >> 5;
    const int lane15 = lane & 15;
    const int hi8    = (lane >= 16) ? 8 : 0;
    const int koffA  = (lane >= 16) ? 8 : 0;
    const int khalfB = (lane >= 16) ? 16 : 0;
    const int rBaseG = blockIdx.x * 64;              // global row base
    const int b      = rBaseG / L_SEQ;
    const int l0     = rBaseG % L_SEQ;
    const float scale = 0.04419417382415922f;        // 1/sqrt(512)

    for (int i = tid; i < 64 * D_DIM; i += 128) O[i] = 0.f;
    __syncthreads();

    float m_i[8], l_i[8];
    #pragma unroll
    for (int r = 0; r < 8; ++r) { m_i[r] = -1e30f; l_i[r] = 0.f; }

    const bf16* Qrow = Qb + (size_t)(rBaseG + wave * 16 + lane15) * D_DIM;
    bf16* myP = Pst + wave * 16 * 32;

    for (int n0 = 0; n0 < L_SEQ; n0 += 32) {
        // ---- S = Q K^T over D=512 ----
        v8f s0 = {}, s1 = {};
        const bf16* Krow0 = Kb + (size_t)(b * L_SEQ + n0 + lane15) * D_DIM;
        const bf16* Krow1 = Krow0 + (size_t)16 * D_DIM;
        #pragma unroll 4
        for (int dk = 0; dk < D_DIM; dk += 32) {
            v16bf aq = mk_afrag(Qrow + dk, koffA);
            v16bf b0 = mk_bfrag(Krow0 + dk, khalfB);
            v16bf b1 = mk_bfrag(Krow1 + dk, khalfB);
            s0 = wmma_bf16(aq, b0, s0);
            s1 = wmma_bf16(aq, b1, s1);
        }
        // ---- mask + scale + online softmax (rows in half-wave) ----
        float corr_r[8];
        #pragma unroll
        for (int r = 0; r < 8; ++r) {
            int lrow = l0 + wave * 16 + r + hi8;
            const int* arow = adj + ((size_t)b * L_SEQ + lrow) * L_SEQ + n0 + lane15;
            // prefetch next tile's mask row (adj is the HBM-bound stream)
            __builtin_prefetch(arow + 32, 0, 1);
            // finite -1e30 mask: fully-masked rows -> exp(0)=1 everywhere -> uniform (matches ref)
            float v0 = (arow[0]  != 0) ? s0[r] * scale : -1e30f;
            float v1 = (arow[16] != 0) ? s1[r] * scale : -1e30f;
            float mx = fmaxf(v0, v1);
            #pragma unroll
            for (int off = 1; off < 16; off <<= 1)
                mx = fmaxf(mx, __shfl_xor(mx, off, 32));
            float mnew = fmaxf(m_i[r], mx);
            float corr = __expf(m_i[r] - mnew);
            float p0 = __expf(v0 - mnew);
            float p1 = __expf(v1 - mnew);
            float rs = p0 + p1;
            #pragma unroll
            for (int off = 1; off < 16; off <<= 1)
                rs += __shfl_xor(rs, off, 32);
            l_i[r] = l_i[r] * corr + rs;
            m_i[r] = mnew;
            corr_r[r] = corr;
            int m = r + hi8;
            myP[m * 32 + lane15]      = f2bf(p0);
            myP[m * 32 + 16 + lane15] = f2bf(p1);
        }
        if (lane15 == 0) {
            #pragma unroll
            for (int r = 0; r < 8; ++r) corrS[wave * 16 + r + hi8] = corr_r[r];
        }
        // ---- rescale O rows (wave-private; same-wave LDS is in-order) ----
        float* Ow = O + wave * 16 * D_DIM;
        #pragma unroll 1
        for (int rr = 0; rr < 16; ++rr) {
            float cf = corrS[wave * 16 + rr];
            if (cf != 1.0f) {
                for (int j = lane; j < D_DIM; j += 32) Ow[rr * D_DIM + j] *= cf;
            }
        }
        // ---- O += P @ V (B-frag gathered from transposed V) ----
        v16bf ap = mk_afrag(myP + lane15 * 32, koffA);
        #pragma unroll 1
        for (int d0 = 0; d0 < D_DIM; d0 += 16) {
            const bf16* vrow = Vt + (size_t)(d0 + lane15) * NROWS + b * L_SEQ + n0;
            v16bf bv = mk_bfrag(vrow, khalfB);
            float* Oc = O + (wave * 16 + hi8) * D_DIM + d0 + lane15;
            v8f c;
            #pragma unroll
            for (int r = 0; r < 8; ++r) c[r] = Oc[r * D_DIM];
            c = wmma_bf16(ap, bv, c);
            #pragma unroll
            for (int r = 0; r < 8; ++r) Oc[r * D_DIM] = c[r];
        }
    }

    if (lane15 == 0) {
        #pragma unroll
        for (int r = 0; r < 8; ++r) invlS[wave * 16 + r + hi8] = 1.0f / l_i[r];
    }
    __syncthreads();
    for (int i = tid; i < 64 * D_DIM; i += 128) {
        int row = i >> 9;
        Ob[(size_t)(rBaseG + row) * D_DIM + (i & 511)] = f2bf(O[i] * invlS[row]);
    }
}

// ---------------- Kernel 3: y = x + attn_out @ Wo + bo (f32 out) ----------------
__global__ __launch_bounds__(256) void out_gemm(
    const bf16* __restrict__ Ob, const float* __restrict__ Wo,
    const float* __restrict__ x, const float* __restrict__ bo,
    float* __restrict__ Y)
{
    __shared__ bf16 As[128 * 32];
    __shared__ bf16 Bs[128 * 32];
    const int rBase = blockIdx.x * 128;
    const int nBase = blockIdx.y * 128;
    const int tid   = threadIdx.x;
    const int lane  = tid & 31;
    const int lane15 = lane & 15;
    const int wave  = tid >> 5;
    const int waveM = wave >> 1, waveN = wave & 1;
    const int koffA  = (lane >= 16) ? 8 : 0;
    const int khalfB = (lane >= 16) ? 16 : 0;

    v8f zero = {};
    v8f acc[2][4];
    #pragma unroll
    for (int i = 0; i < 2; ++i)
        #pragma unroll
        for (int j = 0; j < 4; ++j) acc[i][j] = zero;

    for (int dk = 0; dk < D_DIM; dk += 32) {
        __syncthreads();
        { // stage attn-out tile (already bf16), b128 copies
            int row = tid >> 1, half = tid & 1;
            const bf16* src = Ob + (size_t)(rBase + row) * D_DIM + dk + half * 16;
            *(v8bf*)(As + row * 32 + half * 16)     = *(const v8bf*)src;
            *(v8bf*)(As + row * 32 + half * 16 + 8) = *(const v8bf*)(src + 8);
        }
        { // stage Wo tile transposed
            int k = tid >> 3, seg = tid & 7;
            const float* src = Wo + (size_t)(dk + k) * D_DIM + nBase + seg * 16;
            #pragma unroll
            for (int j = 0; j < 16; ++j) Bs[(seg * 16 + j) * 32 + k] = f2bf(src[j]);
        }
        __syncthreads();

        v16bf a[2], bb[4];
        #pragma unroll
        for (int i = 0; i < 2; ++i)
            a[i] = mk_afrag(As + (waveM * 32 + i * 16 + lane15) * 32, koffA);
        #pragma unroll
        for (int j = 0; j < 4; ++j)
            bb[j] = mk_bfrag(Bs + (waveN * 64 + j * 16 + lane15) * 32, khalfB);
        #pragma unroll
        for (int i = 0; i < 2; ++i)
            #pragma unroll
            for (int j = 0; j < 4; ++j)
                acc[i][j] = wmma_bf16(a[i], bb[j], acc[i][j]);
    }

    const int hi8 = (lane >= 16) ? 8 : 0;
    #pragma unroll
    for (int i = 0; i < 2; ++i)
        #pragma unroll
        for (int j = 0; j < 4; ++j) {
            int col = nBase + waveN * 64 + j * 16 + lane15;
            #pragma unroll
            for (int r = 0; r < 8; ++r) {
                int row = rBase + waveM * 32 + i * 16 + r + hi8;
                Y[(size_t)row * D_DIM + col] =
                    acc[i][j][r] + x[(size_t)row * D_DIM + col] + bo[col];
            }
        }
}

// ---------------- Kernel 4: LayerNorm over D=512, one wave per row ----------------
__global__ __launch_bounds__(256) void ln_kernel(
    const float* __restrict__ Y, const float* __restrict__ gamma,
    const float* __restrict__ beta, float* __restrict__ out)
{
    const int wave = threadIdx.x >> 5, lane = threadIdx.x & 31;
    const int row = blockIdx.x * 8 + wave;
    const float* y = Y + (size_t)row * D_DIM;
    float v[16];
    float s = 0.f, sq = 0.f;
    #pragma unroll
    for (int j = 0; j < 16; ++j) {
        v[j] = y[lane + j * 32];
        s += v[j]; sq += v[j] * v[j];
    }
    #pragma unroll
    for (int off = 1; off < 32; off <<= 1) {
        s  += __shfl_xor(s,  off, 32);
        sq += __shfl_xor(sq, off, 32);
    }
    float mu   = s * (1.f / 512.f);
    float var  = sq * (1.f / 512.f) - mu * mu;
    float rstd = rsqrtf(var + 1e-5f);
    #pragma unroll
    for (int j = 0; j < 16; ++j) {
        int c = lane + j * 32;
        out[(size_t)row * D_DIM + c] = (v[j] - mu) * rstd * gamma[c] + beta[c];
    }
}

extern "C" void kernel_launch(void* const* d_in, const int* in_sizes, int n_in,
                              void* d_out, int out_size, void* d_ws, size_t ws_size,
                              hipStream_t stream) {
    (void)in_sizes; (void)n_in; (void)out_size; (void)ws_size;
    const float* x    = (const float*)d_in[0];
    const int*   adj  = (const int*)d_in[1];
    const float* Wq   = (const float*)d_in[2];
    const float* Wk   = (const float*)d_in[3];
    const float* Wv   = (const float*)d_in[4];
    const float* Wo   = (const float*)d_in[5];
    const float* bo   = (const float*)d_in[6];
    const float* gamma = (const float*)d_in[7];
    const float* beta  = (const float*)d_in[8];

    char* ws = (char*)d_ws;
    bf16*  Qb = (bf16*)(ws);
    bf16*  Kb = (bf16*)(ws + (size_t)(16u << 20));
    bf16*  Vt = (bf16*)(ws + (size_t)(32u << 20));
    bf16*  Ob = (bf16*)(ws + (size_t)(48u << 20));
    float* Y  = (float*)(ws + (size_t)(64u << 20));

    qkv_gemm_t<0><<<dim3(128, 4), 256, 0, stream>>>(x, Wq, Qb);
    qkv_gemm_t<1><<<dim3(128, 4), 256, 0, stream>>>(x, Wk, Kb);
    qkv_gemm_t<2><<<dim3(128, 4), 256, 0, stream>>>(x, Wv, Vt);

    size_t smem = (size_t)64 * D_DIM * 4 + 4 * 16 * 32 * 2 + 128 * 4;
    attn_kernel<<<dim3(256), 128, smem, stream>>>(Qb, Kb, Vt, adj, Ob);

    out_gemm<<<dim3(128, 4), 256, 0, stream>>>(Ob, Wo, x, bo, Y);

    ln_kernel<<<dim3(2048), 256, 0, stream>>>(Y, gamma, beta, (float*)d_out);
}
